// our_adaptive_22119081575178
// MI455X (gfx1250) — matrix operand order repaired and checked by reading, not compiled, not run on hardware
//
#include <hip/hip_runtime.h>
#include <hip/hip_bf16.h>

// ---------------------------------------------------------------------------
// Problem constants (from reference)
// ---------------------------------------------------------------------------
#define S_DIM 50000
#define E_DIM 20000
#define K_DIM 128
#define D_DIM 64
#define B_DIM 4096
#define NUM_LAYER 3

typedef __attribute__((ext_vector_type(16))) __bf16 v16bf;
typedef __attribute__((ext_vector_type(8)))  float  v8f;
typedef __attribute__((ext_vector_type(4))) unsigned int u32x4;
typedef __attribute__((ext_vector_type(8))) int i32x8;
typedef __attribute__((ext_vector_type(4))) int i32x4;

__device__ __forceinline__ __bf16 f2bf(float f) {
    unsigned u = __builtin_bit_cast(unsigned, f);
    unsigned r = (u + 0x7FFFu + ((u >> 16) & 1u)) >> 16;   // RNE
    return __builtin_bit_cast(__bf16, (unsigned short)r);
}

__device__ __forceinline__ float sigm(float x) {
    return 1.0f / (1.0f + __expf(-x));
}

// ---------------------------------------------------------------------------
// f32 -> bf16 weight conversion (optionally |w| for PosLinear)
// ---------------------------------------------------------------------------
__global__ void cvt_bf16_kernel(const float* __restrict__ src,
                                __bf16* __restrict__ dst, int n, int doAbs) {
    int i = blockIdx.x * blockDim.x + threadIdx.x;
    if (i >= n) return;
    float v = src[i];
    if (doAbs) v = fabsf(v);
    dst[i] = f2bf(v);
}

// ---------------------------------------------------------------------------
// TDM: DMA a [128 rows x kc bf16] W tile into LDS with padded row stride
// WLD=144 elements. D# built per CDNA5 ISA.
// Issued by one wave; completion via s_wait_tensorcnt.
// ---------------------------------------------------------------------------
#define WLD 144

__device__ __forceinline__ void tdm_load_w_tile(const __bf16* gsrc,
                                                __bf16* lds_dst,
                                                int Kin, int kc, int nrows) {
    unsigned long long ga = (unsigned long long)(size_t)gsrc;
    unsigned lds_off = (unsigned)(size_t)lds_dst;   // addr[31:0] == LDS offset

    // ---- group 0: count=1 | lds_addr | global_addr | type=2 ----
    u32x4 g0;
    g0[0] = 1u;
    g0[1] = lds_off;
    g0[2] = (unsigned)(ga & 0xFFFFFFFFu);
    g0[3] = (unsigned)((ga >> 32) & 0x01FFFFFFu) | (2u << 30);

    // ---- group 1 ----
    // pad LDS rows: kc=128 -> 64dw rows (code5), +8dw (code7) => stride 144
    //               kc=64  -> 32dw rows (code4), +40dw (code39) => stride 144
    const int picode = (kc == 64) ? 4 : 5;
    const int pamt   = (WLD - kc) / 2 - 1;          // pad DWORDs - 1
    i32x8 g1;
    g1[0] = (1 << 16)                 // data_size = 2 bytes
          | (1 << 20)                 // pad_enable
          | (picode << 22) | (pamt << 25);
    g1[1] = (Kin & 0xFFFF) << 16;                         // tensor_dim0 lo
    g1[2] = ((unsigned)Kin >> 16) | ((nrows & 0xFFFF) << 16); // dim0 hi | dim1 lo
    g1[3] = ((unsigned)nrows >> 16) | ((kc & 0xFFFF) << 16);  // dim1 hi | tile_dim0
    g1[4] = 128;                                          // tile_dim1 (rows)
    g1[5] = Kin;                                          // tensor_dim0_stride lo
    g1[6] = 0;
    g1[7] = 0;

    i32x4 z4 = (i32x4){0, 0, 0, 0};
#if __has_include(<hip/amd_detail/amd_gfx1250_TDM.h>)
    i32x8 z8 = (i32x8){0, 0, 0, 0, 0, 0, 0, 0};
    __builtin_amdgcn_tensor_load_to_lds(g0, g1, z4, z4, z8, 0);
#else
    __builtin_amdgcn_tensor_load_to_lds(g0, g1, z4, z4, 0);
#endif
    __builtin_amdgcn_s_wait_tensorcnt(0);
}

// ---------------------------------------------------------------------------
// Generic WMMA GEMM:  C[M, Ntotal] (+)= A_f32[M, Kin] @ Wbf[Ntotal, Kin]^T
//                     (+ bias) (sigmoid if act)
// Block: 256 threads (8 waves). Each wave: 16 rows x 128 cols.
// gridDim.x = ceil(M/128), gridDim.y = Ntotal/128.
// ---------------------------------------------------------------------------
__global__ void __launch_bounds__(256, 1)
wmma_gemm_kernel(const float* __restrict__ A,
                 const __bf16* __restrict__ Wbf,
                 const float* __restrict__ bias,
                 float* __restrict__ C,
                 int M, int Kin, int Ntotal,
                 int beta, int act) {
    __shared__ __align__(32) __bf16 Wl[128 * WLD];

    const int tid    = threadIdx.x;
    const int wave   = tid >> 5;
    const int lane   = tid & 31;
    const int nBase  = blockIdx.y * 128;
    const int rowStrip = blockIdx.x * 128 + wave * 16;

    v8f acc[8];
#pragma unroll
    for (int i = 0; i < 8; ++i)
        acc[i] = (v8f){0.f, 0.f, 0.f, 0.f, 0.f, 0.f, 0.f, 0.f};

    int mrow = rowStrip + (lane & 15);
    if (mrow >= M) mrow = M - 1;
    const int koff = (lane & 16) ? 8 : 0;       // A-matrix lane K sub-offset
    const int kb   = (lane & 16) ? 8 : 0;       // B-matrix lane K sub-offset
    const int nlo  = lane & 15;

    for (int k0 = 0; k0 < Kin; k0 += 128) {
        const int kc = (Kin - k0 < 128) ? (Kin - k0) : 128;

        // Tensor Data Mover: stage W chunk (128 x kc bf16) into padded LDS
        if (tid == 0) {
            tdm_load_w_tile(Wbf + (size_t)nBase * Kin + k0, Wl,
                            Kin, kc, Ntotal - nBase);
        }
        __syncthreads();

        for (int kk = 0; kk < kc; kk += 32) {
            // ---- A fragment (16x32 bf16): two runs of 8 contiguous f32 ----
            const float* arow = A + (size_t)mrow * Kin + (k0 + kk + koff);
            __builtin_prefetch(arow + 32, 0, 3);   // global_prefetch_b8
            float4 t0 = *(const float4*)(arow);
            float4 t1 = *(const float4*)(arow + 4);
            float4 t2 = *(const float4*)(arow + 16);
            float4 t3 = *(const float4*)(arow + 20);
            v16bf a;
            a[0]  = f2bf(t0.x); a[1]  = f2bf(t0.y); a[2]  = f2bf(t0.z); a[3]  = f2bf(t0.w);
            a[4]  = f2bf(t1.x); a[5]  = f2bf(t1.y); a[6]  = f2bf(t1.z); a[7]  = f2bf(t1.w);
            a[8]  = f2bf(t2.x); a[9]  = f2bf(t2.y); a[10] = f2bf(t2.z); a[11] = f2bf(t2.w);
            a[12] = f2bf(t3.x); a[13] = f2bf(t3.y); a[14] = f2bf(t3.z); a[15] = f2bf(t3.w);

            // Two half-groups per k-step: preload 4 B fragments (8x ds_load_b128)
            // then 4 WMMAs. Keeps DS reads pipelined with partial waits while
            // bounding live registers (~32 for B frags) to avoid scratch spills.
#pragma unroll
            for (int h = 0; h < 2; ++h) {
                v16bf bf[4];
#pragma unroll
                for (int q = 0; q < 4; ++q) {
                    const int nt = h * 4 + q;
                    const __bf16* brow = &Wl[(nt * 16 + nlo) * WLD + kk + kb];
#pragma unroll
                    for (int e = 0; e < 16; ++e) bf[q][e] = brow[e < 8 ? e : e + 8];
                }
#pragma unroll
                for (int q = 0; q < 4; ++q) {
                    const int nt = h * 4 + q;
                    acc[nt] = __builtin_amdgcn_wmma_f32_16x16x32_bf16(
                        false, a, false, bf[q], (short)0, acc[nt], false, false);
                }
                __builtin_amdgcn_sched_group_barrier(0x100, 8, 0); // 8 DS reads
                __builtin_amdgcn_sched_group_barrier(0x008, 4, 0); // 4 WMMAs
            }
        }
        __syncthreads();
    }

    // ---- store: D layout -> lane holds col nlo, rows r + 8*(lane>>4) ----
    const int rowHalf = (lane >> 4) << 3;
#pragma unroll
    for (int nt = 0; nt < 8; ++nt) {
        const int col = nBase + nt * 16 + nlo;
#pragma unroll
        for (int r = 0; r < 8; ++r) {
            const int row = rowStrip + rowHalf + r;
            if (row < M) {
                float v = acc[nt][r];
                size_t idx = (size_t)row * Ntotal + col;
                if (beta) v += C[idx];
                if (bias) v += bias[col];
                if (act)  v = sigm(v);
                C[idx] = v;
            }
        }
    }
}

// ---------------------------------------------------------------------------
// Out[i, :] = X[i, :] * d[i]   (128 cols, float4 per thread)
// ---------------------------------------------------------------------------
__global__ void scale_rows_kernel(const float* __restrict__ X,
                                  const float* __restrict__ d,
                                  float* __restrict__ Out, int M) {
    int t = blockIdx.x * blockDim.x + threadIdx.x;   // one float4 each
    if (t >= M * 32) return;
    int row = t >> 5;
    float s = d[row];
    float4 x = ((const float4*)X)[t];
    x.x *= s; x.y *= s; x.z *= s; x.w *= s;
    ((float4*)Out)[t] = x;
}

// ---------------------------------------------------------------------------
// SpMM scatter: Out[rows[e], :] += vals[e] * X[cols[e], :]
// one wave per edge, float4 per lane, f32 atomics into L2-resident Out.
// ---------------------------------------------------------------------------
__global__ void spmm_atomic_kernel(const int* __restrict__ rows,
                                   const int* __restrict__ cols,
                                   const float* __restrict__ vals, int nnz,
                                   const float* __restrict__ X,
                                   float* __restrict__ Out) {
    int t = blockIdx.x * blockDim.x + threadIdx.x;
    int e = t >> 5;
    if (e >= nnz) return;
    int lane = t & 31;
    int r = rows[e];
    int c = cols[e];
    float v = vals[e];
    float4 x = ((const float4*)(X + (size_t)c * K_DIM))[lane];
    float* o = Out + (size_t)r * K_DIM + lane * 4;
    atomicAdd(o + 0, v * x.x);
    atomicAdd(o + 1, v * x.y);
    atomicAdd(o + 2, v * x.z);
    atomicAdd(o + 3, v * x.w);
}

// ---------------------------------------------------------------------------
// Head input
// ---------------------------------------------------------------------------
__global__ void head_input_kernel(const float* __restrict__ stat,
                                  const float* __restrict__ kdiff,
                                  const int* __restrict__ stu_id,
                                  const int* __restrict__ exer_id,
                                  const float* __restrict__ kn_emb,
                                  const float* __restrict__ stu_bias,
                                  const float* __restrict__ e_disc,
                                  float* __restrict__ x) {
    int idx = blockIdx.x * blockDim.x + threadIdx.x;
    if (idx >= B_DIM * K_DIM) return;
    int b = idx >> 7;
    int k = idx & 127;
    int sid = stu_id[b];
    int eid = exer_id[b];
    float st = stat[(size_t)sid * K_DIM + k] + stu_bias[sid];
    float kd = kdiff[(size_t)eid * K_DIM + k];
    float disc = sigm(e_disc[eid]);
    x[idx] = disc * (sigm(st) - sigm(kd)) * kn_emb[idx];
}

// ---------------------------------------------------------------------------
// Final PosLinear (N=1): out[b] = sigm( dot(h2[b], |w3|) + b3 )
// ---------------------------------------------------------------------------
__global__ void final_dot_kernel(const float* __restrict__ h2,
                                 const float* __restrict__ w3,
                                 const float* __restrict__ b3,
                                 float* __restrict__ out) {
    int b = blockIdx.x * (blockDim.x >> 5) + (threadIdx.x >> 5);
    int lane = threadIdx.x & 31;
    if (b >= B_DIM) return;
    float4 h = ((const float4*)(h2 + (size_t)b * K_DIM))[lane];
    float4 w = ((const float4*)w3)[lane];
    float s = h.x * fabsf(w.x) + h.y * fabsf(w.y) +
              h.z * fabsf(w.z) + h.w * fabsf(w.w);
#pragma unroll
    for (int off = 16; off > 0; off >>= 1)
        s += __shfl_xor(s, off, 32);
    if (lane == 0) out[b] = sigm(s + b3[0]);
}

// ---------------------------------------------------------------------------
// Host orchestration
// ---------------------------------------------------------------------------
static inline void* bump(char* base, size_t& off, size_t bytes) {
    void* p = base + off;
    off = (off + bytes + 255) & ~(size_t)255;
    return p;
}

extern "C" void kernel_launch(void* const* d_in, const int* in_sizes, int n_in,
                              void* d_out, int out_size, void* d_ws, size_t ws_size,
                              hipStream_t stream) {
    // ---- inputs (setup_inputs order) ----
    const int*   stu_id   = (const int*)  d_in[0];
    const int*   exer_id  = (const int*)  d_in[1];
    const float* kn_emb   = (const float*)d_in[2];
    const float* stu_emb  = (const float*)d_in[3];
    const float* exer_emb = (const float*)d_in[4];
    const float* kn_base  = (const float*)d_in[5];
    const float* stu_bias = (const float*)d_in[6];
    const float* e_disc   = (const float*)d_in[7];
    const float* W1_w     = (const float*)d_in[8];
    const float* W1_b     = (const float*)d_in[9];
    const float* W0_w     = (const float*)d_in[10];
    const float* W0_b     = (const float*)d_in[11];
    const float* pn1_w    = (const float*)d_in[12];
    const float* pn1_b    = (const float*)d_in[13];
    const float* pn2_w    = (const float*)d_in[14];
    const float* pn2_b    = (const float*)d_in[15];
    const float* pn3_w    = (const float*)d_in[16];
    const float* pn3_b    = (const float*)d_in[17];
    const int*   ui1_r = (const int*)d_in[18]; const int* ui1_c = (const int*)d_in[19];
    const float* ui1_v = (const float*)d_in[20];
    const int*   iu1_r = (const int*)d_in[21]; const int* iu1_c = (const int*)d_in[22];
    const float* iu1_v = (const float*)d_in[23];
    const int*   ui0_r = (const int*)d_in[24]; const int* ui0_c = (const int*)d_in[25];
    const float* ui0_v = (const float*)d_in[26];
    const int*   iu0_r = (const int*)d_in[27]; const int* iu0_c = (const int*)d_in[28];
    const float* iu0_v = (const float*)d_in[29];
    const float* d_i_1 = (const float*)d_in[30];
    const float* d_j_1 = (const float*)d_in[31];
    const float* d_i_0 = (const float*)d_in[32];
    const float* d_j_0 = (const float*)d_in[33];

    const int nnz1 = in_sizes[20];
    const int nnz0 = in_sizes[26];

    // ---- workspace carve-out ----
    char* ws = (char*)d_ws;
    size_t off = 0;
    float* statA  = (float*)bump(ws, off, (size_t)S_DIM * K_DIM * 4);
    float* statB  = (float*)bump(ws, off, (size_t)S_DIM * K_DIM * 4);
    float* kdiffA = (float*)bump(ws, off, (size_t)E_DIM * K_DIM * 4);
    float* kdiffB = (float*)bump(ws, off, (size_t)E_DIM * K_DIM * 4);
    float* U1 = (float*)bump(ws, off, (size_t)S_DIM * K_DIM * 4);
    float* U0 = (float*)bump(ws, off, (size_t)S_DIM * K_DIM * 4);
    float* I1 = (float*)bump(ws, off, (size_t)E_DIM * K_DIM * 4);
    float* I0 = (float*)bump(ws, off, (size_t)E_DIM * K_DIM * 4);
    float* xb = (float*)bump(ws, off, (size_t)B_DIM * K_DIM * 4);
    float* h1 = (float*)bump(ws, off, (size_t)B_DIM * 256 * 4);
    float* h2 = (float*)bump(ws, off, (size_t)B_DIM * K_DIM * 4);
    __bf16* knb  = (__bf16*)bump(ws, off, (size_t)K_DIM * D_DIM * 2);
    __bf16* W1bf = (__bf16*)bump(ws, off, (size_t)K_DIM * K_DIM * 2);
    __bf16* W0bf = (__bf16*)bump(ws, off, (size_t)K_DIM * K_DIM * 2);
    __bf16* p1bf = (__bf16*)bump(ws, off, (size_t)256 * K_DIM * 2);
    __bf16* p2bf = (__bf16*)bump(ws, off, (size_t)K_DIM * 256 * 2);

    // ---- weight conversion (bf16; |w| for PosLinear) ----
    {
        int n;
        n = K_DIM * D_DIM;
        cvt_bf16_kernel<<<(n + 255) / 256, 256, 0, stream>>>(kn_base, knb, n, 0);
        n = K_DIM * K_DIM;
        cvt_bf16_kernel<<<(n + 255) / 256, 256, 0, stream>>>(W1_w, W1bf, n, 0);
        cvt_bf16_kernel<<<(n + 255) / 256, 256, 0, stream>>>(W0_w, W0bf, n, 0);
        n = 256 * K_DIM;
        cvt_bf16_kernel<<<(n + 255) / 256, 256, 0, stream>>>(pn1_w, p1bf, n, 1);
        cvt_bf16_kernel<<<(n + 255) / 256, 256, 0, stream>>>(pn2_w, p2bf, n, 1);
    }

    const dim3 blk(256);
    const dim3 gS((S_DIM + 127) / 128, 1);
    const dim3 gE((E_DIM + 127) / 128, 1);

    // ---- initial projections onto knowledge space (Kin = 64) ----
    wmma_gemm_kernel<<<gS, blk, 0, stream>>>(stu_emb,  knb, nullptr, statA,
                                             S_DIM, D_DIM, K_DIM, 0, 0);
    wmma_gemm_kernel<<<gE, blk, 0, stream>>>(exer_emb, knb, nullptr, kdiffA,
                                             E_DIM, D_DIM, K_DIM, 0, 0);

    // ---- GCN layers ----
    float* stat  = statA;  float* statN  = statB;
    float* kdiff = kdiffA; float* kdiffN = kdiffB;
    for (int l = 0; l < NUM_LAYER; ++l) {
        scale_rows_kernel<<<(S_DIM * 32 + 255) / 256, blk, 0, stream>>>(stat,  d_i_1, U1, S_DIM);
        scale_rows_kernel<<<(S_DIM * 32 + 255) / 256, blk, 0, stream>>>(stat,  d_i_0, U0, S_DIM);
        scale_rows_kernel<<<(E_DIM * 32 + 255) / 256, blk, 0, stream>>>(kdiff, d_j_1, I1, E_DIM);
        scale_rows_kernel<<<(E_DIM * 32 + 255) / 256, blk, 0, stream>>>(kdiff, d_j_0, I0, E_DIM);

        spmm_atomic_kernel<<<((size_t)nnz1 * 32 + 255) / 256, blk, 0, stream>>>(
            ui1_r, ui1_c, ui1_v, nnz1, kdiff, U1);
        spmm_atomic_kernel<<<((size_t)nnz1 * 32 + 255) / 256, blk, 0, stream>>>(
            iu1_r, iu1_c, iu1_v, nnz1, stat, I1);
        spmm_atomic_kernel<<<((size_t)nnz0 * 32 + 255) / 256, blk, 0, stream>>>(
            ui0_r, ui0_c, ui0_v, nnz0, kdiff, U0);
        spmm_atomic_kernel<<<((size_t)nnz0 * 32 + 255) / 256, blk, 0, stream>>>(
            iu0_r, iu0_c, iu0_v, nnz0, stat, I0);

        wmma_gemm_kernel<<<gS, blk, 0, stream>>>(U1, W1bf, W1_b, statN,
                                                 S_DIM, K_DIM, K_DIM, 0, 0);
        wmma_gemm_kernel<<<gS, blk, 0, stream>>>(U0, W0bf, W0_b, statN,
                                                 S_DIM, K_DIM, K_DIM, 1, 0);
        wmma_gemm_kernel<<<gE, blk, 0, stream>>>(I1, W1bf, W1_b, kdiffN,
                                                 E_DIM, K_DIM, K_DIM, 0, 0);
        wmma_gemm_kernel<<<gE, blk, 0, stream>>>(I0, W0bf, W0_b, kdiffN,
                                                 E_DIM, K_DIM, K_DIM, 1, 0);

        float* t;
        t = stat;  stat  = statN;  statN  = t;
        t = kdiff; kdiff = kdiffN; kdiffN = t;
    }

    // ---- diagnosis head ----
    head_input_kernel<<<(B_DIM * K_DIM + 255) / 256, blk, 0, stream>>>(
        stat, kdiff, stu_id, exer_id, kn_emb, stu_bias, e_disc, xb);

    const dim3 gB1((B_DIM + 127) / 128, 2);   // N = 256
    const dim3 gB2((B_DIM + 127) / 128, 1);   // N = 128
    wmma_gemm_kernel<<<gB1, blk, 0, stream>>>(xb, p1bf, pn1_b, h1,
                                              B_DIM, K_DIM, 256, 0, 1);
    wmma_gemm_kernel<<<gB2, blk, 0, stream>>>(h1, p2bf, pn2_b, h2,
                                              B_DIM, 256, K_DIM, 0, 1);

    final_dot_kernel<<<(B_DIM + 7) / 8, blk, 0, stream>>>(
        h2, pn3_w, pn3_b, (float*)d_out);
}
